// TransformerBlock_19877108646234
// MI455X (gfx1250) — matrix operand order repaired and checked by reading, not compiled, not run on hardware
//
#include <hip/hip_runtime.h>
#include <stdint.h>

// ---------------------------------------------------------------- constants
#define BB   2
#define SS   2048
#define DD   1024
#define HH   16
#define HDD  64

typedef __attribute__((ext_vector_type(16))) __bf16 bf16x16;
typedef __attribute__((ext_vector_type(8)))  float  f32x8;

union Frag { uint32_t u[8]; bf16x16 v; };

static __device__ __forceinline__ uint16_t f2bf(float f) {
  uint32_t u = __builtin_bit_cast(uint32_t, f);
  u += 0x7fffu + ((u >> 16) & 1u);           // round-to-nearest-even
  return (uint16_t)(u >> 16);
}

static __device__ __forceinline__ f32x8 wmma_bf16(const Frag& a, const Frag& b, f32x8 c) {
  return __builtin_amdgcn_wmma_f32_16x16x32_bf16(false, a.v, false, b.v,
                                                 (short)0, c, false, false);
}

// async global->LDS copy of 16 bytes per lane (CDNA5 ASYNCcnt path)
static __device__ __forceinline__ void async_b128(uint32_t lds_addr, uint32_t goff,
                                                  uint64_t base) {
  asm volatile("global_load_async_to_lds_b128 %0, %1, %2"
               :: "v"(lds_addr), "v"(goff), "s"(base) : "memory");
}
static __device__ __forceinline__ void wait_async0() {
  asm volatile("s_wait_asynccnt 0x0" ::: "memory");
}
static __device__ __forceinline__ uint32_t lds_off(const void* p) {
  return (uint32_t)(uintptr_t)p;   // low 32 bits of flat shared addr = LDS offset
}

// -------------------------------------------- fp32 -> bf16 pair-packed weights
// dst[(k>>1)*N + n] = { bf16(src[(k+1)*N+n]) , bf16(src[k*N+n]) }  (hi,lo)
__global__ void cvt_pack_k(const float* __restrict__ src, uint32_t* __restrict__ dst,
                           int K, int N) {
  int i = blockIdx.x * 256 + threadIdx.x;
  int total = (K >> 1) * N;
  if (i < total) {
    int pr = i / N, n = i - pr * N;
    uint32_t lo = f2bf(src[(size_t)(2 * pr) * N + n]);
    uint32_t hi = f2bf(src[(size_t)(2 * pr + 1) * N + n]);
    dst[i] = lo | (hi << 16);
  }
}

// ---------------------------------------------------------------- layernorm -> bf16
__global__ __launch_bounds__(256) void ln_bf16_k(const float* __restrict__ x,
                                                 const float* __restrict__ g,
                                                 const float* __restrict__ bta,
                                                 uint16_t* __restrict__ out) {
  __shared__ float s1[256], s2[256];
  __shared__ float smu, srs;
  int row = blockIdx.x, tid = threadIdx.x;
  const float* xr = x + (size_t)row * DD;
  float v[4], sum = 0.f, sq = 0.f;
#pragma unroll
  for (int j = 0; j < 4; ++j) { v[j] = xr[tid + j * 256]; sum += v[j]; sq += v[j] * v[j]; }
  s1[tid] = sum; s2[tid] = sq; __syncthreads();
  for (int off = 128; off > 0; off >>= 1) {
    if (tid < off) { s1[tid] += s1[tid + off]; s2[tid] += s2[tid + off]; }
    __syncthreads();
  }
  if (tid == 0) {
    float mu = s1[0] * (1.0f / DD);
    float var = s2[0] * (1.0f / DD) - mu * mu;
    smu = mu; srs = rsqrtf(var + 1e-5f);
  }
  __syncthreads();
  float mu = smu, rs = srs;
#pragma unroll
  for (int j = 0; j < 4; ++j) {
    int c = tid + j * 256;
    out[(size_t)row * DD + c] = f2bf((v[j] - mu) * rs * g[c] + bta[c]);
  }
}

// ---------------------------------------------------------------- WMMA GEMM
// C[M,N] = A[M,K](bf16, row-major) @ W(pair-packed dwords, K/2 x N) + bias.
// Double-buffered LDS staged with async global->LDS B128 copies.
// OP 0: QKV scatter   OP 1: fp32 out = val + resid   OP 2: bf16 out = GELU(val)
template <int OP>
__global__ __launch_bounds__(256) void gemm_wmma_k(
    const uint16_t* __restrict__ A, const uint32_t* __restrict__ W,
    const float* __restrict__ bias, int M, int N, int K,
    float* __restrict__ outF, uint16_t* __restrict__ outBF,
    const float* __restrict__ resid,
    uint16_t* __restrict__ qOut, uint16_t* __restrict__ kOut,
    uint16_t* __restrict__ vTOut) {
  __shared__ uint32_t ldsA[2][128 * 20];   // 128 rows x 16 dwords, stride 20 (16B-aligned, bank-safe)
  __shared__ uint32_t ldsB[2][16 * 128];   // 16 pair-rows x 128 cols

  const int tid = threadIdx.x, lane = tid & 31, wid = tid >> 5;
  const int half = lane >> 4, l15 = lane & 15;
  const int mbase = blockIdx.y * 128, nbase = blockIdx.x * 128;
  const int mw = (wid & 1) * 64, nw = (wid >> 1) * 32;
  const int Kd = K >> 1;
  const uint64_t Abase = (uint64_t)(uintptr_t)A;
  const uint64_t Wbase = (uint64_t)(uintptr_t)W;

  f32x8 acc[4][2];
#pragma unroll
  for (int mt = 0; mt < 4; ++mt)
#pragma unroll
    for (int nt = 0; nt < 2; ++nt) acc[mt][nt] = (f32x8)0.0f;

  // issue one 32-k stage of async copies into buffer `buf`
  auto issue_stage = [&](int kb, int buf) {
#pragma unroll
    for (int i = 0; i < 2; ++i) {               // A tile: 128x16 dwords
      int c = tid + (i << 8);                   // 0..511 chunks of 4 dwords
      int r = c >> 2, cd4 = (c & 3) << 2;
      uint32_t go = (uint32_t)((((mbase + r) * Kd + (kb >> 1) + cd4)) << 2);
      async_b128(lds_off(&ldsA[buf][r * 20 + cd4]), go, Abase);
    }
#pragma unroll
    for (int i = 0; i < 2; ++i) {               // B tile: 16x128 dwords (pair-packed)
      int c = tid + (i << 8);
      int pr = c >> 5, n4 = (c & 31) << 2;
      uint32_t go = (uint32_t)(((((kb >> 1) + pr) * N + nbase + n4)) << 2);
      async_b128(lds_off(&ldsB[buf][pr * 128 + n4]), go, Wbase);
    }
  };

  issue_stage(0, 0);
  const int nIt = K >> 5;
  for (int it = 0; it < nIt; ++it) {
    const int buf = it & 1;
    wait_async0();
    __syncthreads();
    if (it + 1 < nIt) issue_stage((it + 1) << 5, buf ^ 1);

    Frag af[4], bf[2];
#pragma unroll
    for (int mt = 0; mt < 4; ++mt) {
      int m = mw + mt * 16 + l15;
#pragma unroll
      for (int v = 0; v < 8; ++v) {
        int kd = (half << 3) + v;
        af[mt].u[v] = ldsA[buf][m * 20 + kd];
      }
    }
#pragma unroll
    for (int nt = 0; nt < 2; ++nt) {
      int n = nw + nt * 16 + l15;
#pragma unroll
      for (int v = 0; v < 8; ++v) {
        int pr = (half << 3) + v;
        bf[nt].u[v] = ldsB[buf][pr * 128 + n];
      }
    }
#pragma unroll
    for (int mt = 0; mt < 4; ++mt)
#pragma unroll
      for (int nt = 0; nt < 2; ++nt)
        acc[mt][nt] = wmma_bf16(af[mt], bf[nt], acc[mt][nt]);
  }

  // epilogue: element (row = v + 8*half, col = l15) within each 16x16 tile
#pragma unroll
  for (int mt = 0; mt < 4; ++mt)
#pragma unroll
    for (int nt = 0; nt < 2; ++nt)
#pragma unroll
      for (int i = 0; i < 8; ++i) {
        int row = mbase + mw + mt * 16 + i + (half << 3);
        int col = nbase + nw + nt * 16 + l15;
        float val = acc[mt][nt][i] + bias[col];
        if (OP == 0) {
          int which = col >> 10;          // 0=q 1=k 2=v
          int d = col & (DD - 1);
          int h = d >> 6, hd = d & 63;
          int b = row >> 11, s = row & (SS - 1);
          int bh = b * HH + h;
          uint16_t bv = f2bf(val);
          if (which == 0)      qOut[((size_t)(bh * SS + s)) * HDD + hd] = bv;
          else if (which == 1) kOut[((size_t)(bh * SS + s)) * HDD + hd] = bv;
          else                 vTOut[((size_t)(bh * HDD + hd)) * SS + s] = bv;
        } else if (OP == 1) {
          size_t o = (size_t)row * N + col;
          outF[o] = val + resid[o];
        } else {
          float gl = 0.5f * val * (1.0f + erff(val * 0.70710678118654752f));
          outBF[(size_t)row * N + col] = f2bf(gl);
        }
      }
}

// ---------------------------------------------------------------- flash attention
// One wave per (bh, 16-query tile). Scores & PV via bf16 WMMA, online softmax.
__global__ __launch_bounds__(32) void attn_flash_k(
    const uint16_t* __restrict__ qb, const uint16_t* __restrict__ kb,
    const uint16_t* __restrict__ vT, const unsigned char* __restrict__ pad,
    uint16_t* __restrict__ attnOut) {
  __shared__ uint32_t ldsP[16 * 17];
  const int lane = threadIdx.x;
  const int half = lane >> 4, l15 = lane & 15;
  const int bh = blockIdx.y, b = bh >> 4, h = bh & 15;
  const int q0 = blockIdx.x * 16;

  const uint32_t* q32 = (const uint32_t*)qb;
  const uint32_t* k32 = (const uint32_t*)kb;
  const uint32_t* v32 = (const uint32_t*)vT;

  Frag qf[2];
#pragma unroll
  for (int c = 0; c < 2; ++c)
#pragma unroll
    for (int v = 0; v < 8; ++v) {
      int kd = c * 16 + (half << 3) + v;
      qf[c].u[v] = q32[((size_t)(bh * SS) + q0 + l15) * 32 + kd];
    }

  f32x8 o[4];
#pragma unroll
  for (int nt = 0; nt < 4; ++nt) o[nt] = (f32x8)0.0f;
  float rmax[8], rsum[8];
#pragma unroll
  for (int i = 0; i < 8; ++i) { rmax[i] = -1e30f; rsum[i] = 0.f; }

  uint16_t* p16 = (uint16_t*)ldsP;

  for (int kbase = 0; kbase < q0 + 16; kbase += 32) {
    __syncthreads();
    // ---- scores: two 16x16 tiles over HD=64 (two k-steps each)
    f32x8 sc[2];
#pragma unroll
    for (int t = 0; t < 2; ++t) {
      sc[t] = (f32x8)0.0f;
#pragma unroll
      for (int c = 0; c < 2; ++c) {
        Frag kf;
#pragma unroll
        for (int v = 0; v < 8; ++v) {
          int pr = (half << 3) + v;
          kf.u[v] = k32[((size_t)(bh * SS) + kbase + t * 16 + l15) * 32 + c * 16 + pr];
        }
        sc[t] = wmma_bf16(qf[c], kf, sc[t]);
      }
    }
    // ---- mask + online softmax stats
    int kk0 = kbase + l15, kk1 = kbase + 16 + l15;
    int pm0 = pad[b * SS + kk0] ? 1 : 0;
    int pm1 = pad[b * SS + kk1] ? 1 : 0;
    float s0[8], s1[8], tm[8];
#pragma unroll
    for (int i = 0; i < 8; ++i) {
      int qr = q0 + i + (half << 3);
      s0[i] = (kk0 > qr || pm0) ? -1e30f : sc[0][i] * 0.125f;
      s1[i] = (kk1 > qr || pm1) ? -1e30f : sc[1][i] * 0.125f;
      tm[i] = fmaxf(s0[i], s1[i]);
    }
    for (int mm = 1; mm < 16; mm <<= 1)
#pragma unroll
      for (int i = 0; i < 8; ++i) tm[i] = fmaxf(tm[i], __shfl_xor(tm[i], mm, 32));

    float corr[8], part[8];
#pragma unroll
    for (int i = 0; i < 8; ++i) {
      float nm = fmaxf(rmax[i], tm[i]);
      corr[i] = expf(rmax[i] - nm);
      rmax[i] = nm;
      float x0 = expf(s0[i] - nm);
      float x1 = expf(s1[i] - nm);
      part[i] = x0 + x1;
      int m = i + (half << 3);
      p16[m * 34 + l15] = f2bf(x0);
      p16[m * 34 + 16 + l15] = f2bf(x1);
    }
    for (int mm = 1; mm < 16; mm <<= 1)
#pragma unroll
      for (int i = 0; i < 8; ++i) part[i] += __shfl_xor(part[i], mm, 32);
#pragma unroll
    for (int i = 0; i < 8; ++i) rsum[i] = rsum[i] * corr[i] + part[i];
#pragma unroll
    for (int nt = 0; nt < 4; ++nt)
#pragma unroll
      for (int i = 0; i < 8; ++i) o[nt][i] *= corr[i];

    __syncthreads();
    // ---- P (A-frag via LDS transpose) @ V (B-frag from vT, pair-packed dwords)
    Frag pf;
#pragma unroll
    for (int v = 0; v < 8; ++v) {
      int kd = (half << 3) + v;
      pf.u[v] = ldsP[l15 * 17 + kd];
    }
#pragma unroll
    for (int nt = 0; nt < 4; ++nt) {
      Frag vf;
#pragma unroll
      for (int v = 0; v < 8; ++v) {
        int pr = (half << 3) + v;
        int hd = nt * 16 + l15;
        vf.u[v] = v32[((size_t)(bh * HDD + hd)) * (SS / 2) + (kbase >> 1) + pr];
      }
      o[nt] = wmma_bf16(pf, vf, o[nt]);
    }
  }

  // ---- normalize + write token-major bf16
#pragma unroll
  for (int nt = 0; nt < 4; ++nt)
#pragma unroll
    for (int i = 0; i < 8; ++i) {
      int qr = q0 + i + (half << 3);
      float val = o[nt][i] / rsum[i];
      attnOut[((size_t)(b * SS + qr)) * DD + h * HDD + nt * 16 + l15] = f2bf(val);
    }
}

// ---------------------------------------------------------------- host launch
extern "C" void kernel_launch(void* const* d_in, const int* in_sizes, int n_in,
                              void* d_out, int out_size, void* d_ws, size_t ws_size,
                              hipStream_t stream) {
  const float* x        = (const float*)d_in[0];
  const unsigned char* pad = (const unsigned char*)d_in[1];
  const float* qkv_w    = (const float*)d_in[2];
  const float* qkv_b    = (const float*)d_in[3];
  const float* out_w    = (const float*)d_in[4];
  const float* out_b    = (const float*)d_in[5];
  const float* ln1_g    = (const float*)d_in[6];
  const float* ln1_b    = (const float*)d_in[7];
  const float* ff1_w    = (const float*)d_in[8];
  const float* ff1_b    = (const float*)d_in[9];
  const float* ff2_w    = (const float*)d_in[10];
  const float* ff2_b    = (const float*)d_in[11];
  const float* ln2_g    = (const float*)d_in[12];
  const float* ln2_b    = (const float*)d_in[13];

  char* base = (char*)d_ws;
  uint32_t* qkvw_p  = (uint32_t*)(base + 0);          // 6291456
  uint32_t* outw_p  = (uint32_t*)(base + 6291456);    // 2097152
  uint32_t* ff1w_p  = (uint32_t*)(base + 8388608);    // 8388608
  uint32_t* ff2w_p  = (uint32_t*)(base + 16777216);   // 8388608
  uint16_t* hbf     = (uint16_t*)(base + 25165824);   // 8388608 (LN1 out, reused for LN2 out)
  uint16_t* qb      = (uint16_t*)(base + 33554432);   // 8388608
  uint16_t* kbuf    = (uint16_t*)(base + 41943040);   // 8388608
  uint16_t* vT      = (uint16_t*)(base + 50331648);   // 8388608
  uint16_t* attnb   = (uint16_t*)(base + 58720256);   // 8388608
  float*    x1      = (float*)   (base + 67108864);   // 16777216
  uint16_t* h3      = qb;  // reuse q/k/vT/attn region (33554432 B) for FF1 output

  const int M = BB * SS;  // 4096

  // weight conversion + pair-packing
  { int n = 512 * 3072;  cvt_pack_k<<<(n + 255) / 256, 256, 0, stream>>>(qkv_w, qkvw_p, DD, 3 * DD); }
  { int n = 512 * 1024;  cvt_pack_k<<<(n + 255) / 256, 256, 0, stream>>>(out_w, outw_p, DD, DD); }
  { int n = 512 * 4096;  cvt_pack_k<<<(n + 255) / 256, 256, 0, stream>>>(ff1_w, ff1w_p, DD, 4 * DD); }
  { int n = 2048 * 1024; cvt_pack_k<<<(n + 255) / 256, 256, 0, stream>>>(ff2_w, ff2w_p, 4 * DD, DD); }

  // LN1
  ln_bf16_k<<<M, 256, 0, stream>>>(x, ln1_g, ln1_b, hbf);

  // QKV: [4096,1024] x [1024,3072]
  gemm_wmma_k<0><<<dim3(3 * DD / 128, M / 128), 256, 0, stream>>>(
      hbf, qkvw_p, qkv_b, M, 3 * DD, DD,
      nullptr, nullptr, nullptr, qb, kbuf, vT);

  // attention
  attn_flash_k<<<dim3(SS / 16, BB * HH), 32, 0, stream>>>(qb, kbuf, vT, pad, attnb);

  // out-proj + residual -> x1
  gemm_wmma_k<1><<<dim3(DD / 128, M / 128), 256, 0, stream>>>(
      attnb, outw_p, out_b, M, DD, DD,
      x1, nullptr, x, nullptr, nullptr, nullptr);

  // LN2
  ln_bf16_k<<<M, 256, 0, stream>>>(x1, ln2_g, ln2_b, hbf);

  // FF1 + GELU -> h3 (bf16)
  gemm_wmma_k<2><<<dim3(4 * DD / 128, M / 128), 256, 0, stream>>>(
      hbf, ff1w_p, ff1_b, M, 4 * DD, DD,
      nullptr, h3, nullptr, nullptr, nullptr, nullptr);

  // FF2 + residual -> d_out (fp32)
  gemm_wmma_k<1><<<dim3(DD / 128, M / 128), 256, 0, stream>>>(
      h3, ff2w_p, ff2_b, M, DD, 4 * DD,
      (float*)d_out, nullptr, x1, nullptr, nullptr, nullptr);
}
